// RvT_78666620994150
// MI455X (gfx1250) — compile-verified, matrix operand order checked by reading, not run on hardware
//
#include <hip/hip_runtime.h>
#include <cstdint>

// ---------------------------------------------------------------------------
// Problem constants (from reference):
//   N=2, CIN=256, C=512, G=8, CG=64, K=3, H=W=48 -> HW=2304
//   k/v spatial: 46x46 -> D=2116.  xh has 2C=1024 channels, 2CG=128 per group.
// ---------------------------------------------------------------------------
#define NB    2
#define CINCH 256
#define CCH   512
#define GRP   8
#define CG    64
#define HH    48
#define WW    48
#define HWP   2304          // q / gate pixels
#define ODIM  46
#define DKV   2116          // k/v pixels
#define K9    1152          // 2*CG*3*3 (grouped conv GEMM K)
#define KS3   36            // K9/32
#define KSP   24            // proj GEMM K = 768 -> 24 steps

typedef __bf16 bf16_t;
typedef __attribute__((ext_vector_type(16))) __bf16 v16bf;
typedef __attribute__((ext_vector_type(8)))  float  v8f;
typedef __attribute__((ext_vector_type(4)))  unsigned int v4u;
typedef __attribute__((ext_vector_type(8)))  int    v8i_t;
typedef __attribute__((ext_vector_type(4)))  int    v4i_t;

// ---------------------------------------------------------------------------
// WMMA fragment helpers (CDNA5 16-bit operand layout, CDNA5 ISA):
//  A 16x32: lane = m + 16*khalf, khalf selects K subset {0..7,16..23} vs
//  {8..15,24..31}; within a lane, 16 values packed so one 32B load per lane.
//  B tiles are staged pre-swizzled into the mirrored layout (lane = n+16*khalf)
//  so every operand fetch is a contiguous v16bf (2x ds_load_b128) read.
// ---------------------------------------------------------------------------
__device__ __forceinline__ int frag_pos(int k, int rc) {
  int khalf = (k >> 3) & 1;
  int lane  = rc | (khalf << 4);
  int slot  = (k & 7) | ((k >> 4) << 3);
  return (lane << 4) | slot;          // index inside a 512-element fragment
}
__device__ __forceinline__ int slot_k(int slot, int khalf) {
  return ((slot & 7) | (khalf << 3)) + ((slot >> 3) << 4);
}
__device__ __forceinline__ v16bf load_frag(const bf16_t* base, int lane) {
  return *reinterpret_cast<const v16bf*>(base + (lane << 4));
}
__device__ __forceinline__ v8f wmma_bf16(v16bf a, v16bf b, v8f c) {
  return __builtin_amdgcn_wmma_f32_16x16x32_bf16(false, a, false, b,
                                                 (short)0, c, false, false);
}

// ---------------------------------------------------------------------------
// Tensor Data Mover: 2D tile load Global->LDS (CDNA5 ISA)
//  D# group0: [1:0]count=1 | [63:32]lds_addr | [120:64]global_addr | type=2
//  D# group1: [17:16]data_size(1=2B) | dim0 | dim1 | tile0 | tile1 | dim0_stride
//  Issued from one wave (wave-level op, EXEC ignored); completion tracked on
//  TENSORcnt.  OOB columns (beyond tensor_dim0) read back as zero.
//  This toolchain exposes the 6-arg builtin (clang-23 / therock-10.0): the
//  trailing int32x8 carries the optional descriptor groups (zero => 2D only).
// ---------------------------------------------------------------------------
__device__ __forceinline__ void tdm_load_2d(const void* gptr, unsigned int lds_off,
                                            unsigned int dim0, unsigned int stride0,
                                            unsigned int tile1, unsigned int tile0) {
  unsigned long long ga = (unsigned long long)(uintptr_t)gptr;
  v4u g0;
  g0[0] = 1u;                                               // count=1 (valid, user)
  g0[1] = lds_off;                                          // lds_addr (bytes)
  g0[2] = (unsigned int)ga;                                 // global_addr[31:0]
  g0[3] = (unsigned int)((ga >> 32) & 0x01FFFFFFu) | (2u << 30); // addr[56:32]|type=2
  v8i_t g1;
  g1[0] = (int)(1u << 16);                                  // data_size=2B, mask=0
  g1[1] = (int)((dim0 & 0xFFFFu) << 16);                    // tensor_dim0[15:0]
  g1[2] = (int)((dim0 >> 16) | ((tile1 & 0xFFFFu) << 16));  // dim0 hi | tensor_dim1 lo
  g1[3] = (int)((tile0 & 0xFFFFu) << 16);                   // dim1 hi(0) | tile_dim0
  g1[4] = (int)(tile1 & 0xFFFFu);                           // tile_dim1 | tile_dim2=0
  g1[5] = (int)stride0;                                     // tensor_dim0_stride lo
  g1[6] = 0;                                                // stride hi | dim1_stride
  g1[7] = 0;
  v4i_t z4 = {};
  v8i_t z8 = {};
  __builtin_amdgcn_tensor_load_to_lds(g0, g1, z4, z4, z8, 0);
}
#define LDS_OFF(p) ((unsigned int)(uintptr_t)(p))  // low 32b of LDS aperture addr

// ---------------------------------------------------------------------------
// Weight pre-pack: fp32 row-major [M x Ksrc] -> bf16 fragment-packed
// [(g*mtPerG+mt)*Ksteps + ks][512], with global K offset (for concatenated K).
// ---------------------------------------------------------------------------
__global__ void pack_a_kernel(const float* __restrict__ W, bf16_t* __restrict__ out,
                              int M, int Ksrc, int srcStride, int kOff, int Ktot,
                              int rowsPerGroup) {
  int idx = blockIdx.x * blockDim.x + threadIdx.x;
  if (idx >= M * Ksrc) return;
  int m = idx / Ksrc, k = idx - m * Ksrc;
  int kg = k + kOff;
  int ksteps = Ktot >> 5;
  int mtPerG = rowsPerGroup >> 4;
  int g  = m / rowsPerGroup;
  int mt = (m % rowsPerGroup) >> 4;
  int fi = frag_pos(kg & 31, m & 15);
  out[((((g * mtPerG + mt) * ksteps) + (kg >> 5)) << 9) + fi] =
      (bf16_t)W[m * srcStride + k];
}

// h (fp32 NCHW) -> bf16 slots of interleaved xh  [n][g*128+64+cg][p]
__global__ void h_to_xh_kernel(const float* __restrict__ h, bf16_t* __restrict__ xh) {
  int idx = blockIdx.x * blockDim.x + threadIdx.x;
  if (idx >= NB * CCH * HWP) return;
  int n  = idx / (CCH * HWP);
  int r  = idx - n * CCH * HWP;
  int ch = r / HWP;
  int p  = r - ch * HWP;
  xh[((n * 1024) + (ch >> 6) * 128 + 64 + (ch & 63)) * HWP + p] = (bf16_t)h[idx];
}

// ---------------------------------------------------------------------------
// proj GEMM: x = [W_x | W_ig] * [x_in ; h]  (M=512, K=768, per batch N=2304)
// 8 waves/block, each owns a 16-row tile; shared B tile staged in frag order
// (fp32 -> bf16 conversion happens in the staging pass, so no TDM here).
// Result written (bf16) straight into the x-slots of interleaved xh.
// ---------------------------------------------------------------------------
__global__ void __launch_bounds__(256) proj_gemm_kernel(
    const float* __restrict__ xin, const float* __restrict__ h,
    const bf16_t* __restrict__ packW, bf16_t* __restrict__ xh) {
  __shared__ __align__(32) bf16_t ldsB[512];
  int lane = threadIdx.x & 31, wave = threadIdx.x >> 5;
  int tp = blockIdx.x;                    // 16-pixel tile
  int mt = blockIdx.y * 8 + wave;         // 16-row tile of C (0..31)
  int nb = blockIdx.z;
  v8f acc = {};
  for (int ks = 0; ks < KSP; ++ks) {
    for (int e = threadIdx.x; e < 512; e += 256) {
      int k = e >> 4, nn = e & 15;
      int kk = ks * 32 + k;
      int p  = tp * 16 + nn;
      float v = (kk < CINCH) ? xin[(nb * CINCH + kk) * HWP + p]
                             : h[(nb * CCH + (kk - CINCH)) * HWP + p];
      ldsB[frag_pos(k, nn)] = (bf16_t)v;
    }
    __syncthreads();
    v16bf a = load_frag(packW + ((mt * KSP + ks) << 9), lane);
    v16bf b = load_frag(ldsB, lane);
    acc = wmma_bf16(a, b, acc);
    __syncthreads();
  }
  int p = tp * 16 + (lane & 15);
  int rbase = mt * 16 + ((lane >> 4) << 3);
  for (int r = 0; r < 8; ++r) {
    int m = rbase + r;
    xh[((nb * 1024) + (m >> 6) * 128 + (m & 63)) * HWP + p] = (bf16_t)acc[r];
  }
}

// ---------------------------------------------------------------------------
// Grouped 3x3 conv as implicit GEMM: Out[64, P] = Wg[64,1152] x Im2col[1152,P]
// 8 waves/block = 4 row-tiles x 2 pixel-tiles (block tile 64x32).
// Im2col tiles gathered from xh (bf16) directly into fragment-ordered LDS
// (halo/padding gather is not a rectangular tile, so no TDM here).
// ---------------------------------------------------------------------------
__global__ void __launch_bounds__(256) conv3x3_gemm_kernel(
    const bf16_t* __restrict__ xh, const bf16_t* __restrict__ packW,
    float* __restrict__ out, int pad, int outW, int P) {
  __shared__ __align__(32) bf16_t ldsB[1024];
  int lane = threadIdx.x & 31, wave = threadIdx.x >> 5;
  int tp = blockIdx.x;                    // 32-pixel tile
  int ng = blockIdx.y;                    // n*G + g
  int nb = ng >> 3, g = ng & 7;
  int mt = wave >> 1, nt = wave & 1;
  const bf16_t* xbase = xh + (nb * 1024 + g * 128) * HWP;
  v8f acc = {};
  for (int ks = 0; ks < KS3; ++ks) {
    for (int e = threadIdx.x; e < 1024; e += 256) {
      int ntt = e >> 9, ee = e & 511;
      int k = ee >> 4, nn = ee & 15;
      int kk = ks * 32 + k;               // 0..1151
      int c2 = kk / 9, rs = kk - c2 * 9;
      int p  = tp * 32 + ntt * 16 + nn;
      bf16_t v = (bf16_t)0.f;
      if (p < P) {
        int y  = p / outW, x0 = p - y * outW;
        int iy = y - pad + rs / 3;
        int ix = x0 - pad + (rs % 3);
        if (iy >= 0 && iy < HH && ix >= 0 && ix < WW)
          v = xbase[(c2 * HH + iy) * WW + ix];
      }
      ldsB[(ntt << 9) + frag_pos(k, nn)] = v;
    }
    __syncthreads();
    v16bf a = load_frag(packW + (((g * 4 + mt) * KS3 + ks) << 9), lane);
    v16bf b = load_frag(&ldsB[nt << 9], lane);
    acc = wmma_bf16(a, b, acc);
    __syncthreads();
  }
  int p = tp * 32 + nt * 16 + (lane & 15);
  if (p < P) {
    int rbase = mt * 16 + ((lane >> 4) << 3);
    for (int r = 0; r < 8; ++r)
      out[(ng * CG + rbase + r) * P + p] = acc[r];
  }
}

// q fp32 [ng][64][2304] -> bf16 transposed [ng][2304][64] (A-side of attention)
__global__ void repack_qT_kernel(const float* __restrict__ qf, bf16_t* __restrict__ qT) {
  int idx = blockIdx.x * blockDim.x + threadIdx.x;
  if (idx >= 16 * CG * HWP) return;
  int ng = idx / (CG * HWP);
  int r  = idx - ng * CG * HWP;
  int c  = r / HWP;
  int p  = r - c * HWP;
  qT[(ng * HWP + p) * CG + c] = (bf16_t)qf[idx];
}

__global__ void cvt_f32_bf16_kernel(const float* __restrict__ s,
                                    bf16_t* __restrict__ d, int total) {
  int idx = blockIdx.x * blockDim.x + threadIdx.x;
  if (idx < total) d[idx] = (bf16_t)s[idx];
}

// ---------------------------------------------------------------------------
// Flash attention per (n,g): a = V * softmax(tau * Q^T K)^T, never
// materializing the [2304 x 2116] logits (312 MB -> would be HBM-bound).
// 8 waves/block, one 16-row q-tile each.  K/V are streamed in 32-column
// chunks by the Tensor Data Mover (one 64x32 D# tile each, TENSORcnt),
// re-striped LDS->LDS into WMMA fragment order, then consumed by WMMA.
// Row max/sum kept per lane with shfl_xor reductions over the 16-lane column
// group (wave32).  P tile goes through a per-wave LDS region to be re-striped
// into A-fragment layout for the second WMMA.
// ---------------------------------------------------------------------------
__global__ void __launch_bounds__(256) attention_kernel(
    const bf16_t* __restrict__ qT, const bf16_t* __restrict__ kb,
    const bf16_t* __restrict__ vb, const float* __restrict__ tau,
    bf16_t* __restrict__ aout) {
  __shared__ __align__(32) bf16_t rawK[2048];   // 64 c-rows x 32 d (TDM dest)
  __shared__ __align__(32) bf16_t rawV[2048];   // 64 c-rows x 32 d (TDM dest)
  __shared__ __align__(32) bf16_t ldsK[2048];   // 2 ksteps(c) x 2 dtiles frags
  __shared__ __align__(32) bf16_t ldsV[2048];   // 4 ctiles x (kd=32) frags
  __shared__ __align__(32) bf16_t ldsP[8 * 512];
  int lane = threadIdx.x & 31, wave = threadIdx.x >> 5;
  int ng = blockIdx.y;
  int qt = blockIdx.x * 8 + wave;               // 0..143
  float taug = tau[ng & 7];
  const bf16_t* km = kb + (size_t)ng * CG * DKV;
  const bf16_t* vm = vb + (size_t)ng * CG * DKV;

  // Q A-fragments for this wave's 16 q rows (K = Cg = 64 -> 2 ksteps)
  v16bf qa0, qa1;
  {
    int khalf = lane >> 4;
    const bf16_t* qrow = qT + ((size_t)ng * HWP + qt * 16 + (lane & 15)) * CG;
    for (int i = 0; i < 16; ++i) {
      qa0[i] = qrow[slot_k(i, khalf)];
      qa1[i] = qrow[32 + slot_k(i, khalf)];
    }
  }
  float rmax[8], rsum[8];
  for (int r = 0; r < 8; ++r) { rmax[r] = -3.0e38f; rsum[r] = 0.f; }
  v8f accO[4] = {};

  const int nChunks = (DKV + 31) / 32;          // 67
  for (int ch = 0; ch < nChunks; ++ch) {
    int dbase = ch * 32;
    // DMA this chunk's K/V tiles into LDS (wave-level op, one wave issues).
    // Columns past DKV in the last chunk come back as zeros (TDM OOB rule).
    if (wave == 0) {
      tdm_load_2d(km + dbase, LDS_OFF(rawK), DKV - dbase, DKV, CG, 32);
      tdm_load_2d(vm + dbase, LDS_OFF(rawV), DKV - dbase, DKV, CG, 32);
      __builtin_amdgcn_s_wait_tensorcnt(0);
    }
    __syncthreads();                            // raw tiles visible to all waves
    // re-stripe raw row-major tiles into WMMA B-fragment order
    for (int e = threadIdx.x; e < 2048; e += 256) {   // K: B[k=c][n=d]
      int frag = e >> 9, ee = e & 511;
      int k = ee >> 4, nn = ee & 15;
      int c = ((frag >> 1) << 5) + k;
      int dl = ((frag & 1) << 4) + nn;
      ldsK[(frag << 9) + frag_pos(k, nn)] = rawK[(c << 5) + dl];
    }
    for (int e = threadIdx.x; e < 2048; e += 256) {   // V: B[k=d][n=c]
      int frag = e >> 9, ee = e & 511;
      int k = ee >> 4, nn = ee & 15;
      int c = (frag << 4) + nn;
      ldsV[(frag << 9) + frag_pos(k, nn)] = rawV[(c << 5) + k];
    }
    __syncthreads();

    v8f s0 = {}, s1 = {};
    s0 = wmma_bf16(qa0, load_frag(&ldsK[0 * 512], lane), s0);
    s0 = wmma_bf16(qa1, load_frag(&ldsK[2 * 512], lane), s0);
    s1 = wmma_bf16(qa0, load_frag(&ldsK[1 * 512], lane), s1);
    s1 = wmma_bf16(qa1, load_frag(&ldsK[3 * 512], lane), s1);

    int col0 = dbase + (lane & 15), col1 = col0 + 16;
    float tmax[8];
    for (int r = 0; r < 8; ++r) {
      float v0 = (col0 < DKV) ? s0[r] * taug : -3.0e38f;
      float v1 = (col1 < DKV) ? s1[r] * taug : -3.0e38f;
      s0[r] = v0; s1[r] = v1;
      tmax[r] = fmaxf(v0, v1);
    }
    for (int m = 1; m < 16; m <<= 1)
      for (int r = 0; r < 8; ++r)
        tmax[r] = fmaxf(tmax[r], __shfl_xor(tmax[r], m, 32));
    float scale[8];
    for (int r = 0; r < 8; ++r) {
      float nm = fmaxf(rmax[r], tmax[r]);
      scale[r] = __expf(rmax[r] - nm);
      rmax[r] = nm;
    }
    float tsum[8];
    for (int r = 0; r < 8; ++r) {
      float p0 = (col0 < DKV) ? __expf(s0[r] - rmax[r]) : 0.f;
      float p1 = (col1 < DKV) ? __expf(s1[r] - rmax[r]) : 0.f;
      s0[r] = p0; s1[r] = p1;
      tsum[r] = p0 + p1;
    }
    for (int m = 1; m < 16; m <<= 1)
      for (int r = 0; r < 8; ++r)
        tsum[r] += __shfl_xor(tsum[r], m, 32);
    for (int r = 0; r < 8; ++r)
      rsum[r] = rsum[r] * scale[r] + tsum[r];

    // re-stripe P (C-layout) -> A-fragment layout through per-wave LDS
    bf16_t* pw = &ldsP[wave << 9];
    {
      int half = lane >> 4;
      int kd0 = lane & 15, kd1 = 16 + (lane & 15);
      for (int r = 0; r < 8; ++r) {
        int m = r + (half << 3);
        pw[((m | (((kd0 >> 3) & 1) << 4)) << 4) + ((kd0 & 7) | ((kd0 >> 4) << 3))] = (bf16_t)s0[r];
        pw[((m | (((kd1 >> 3) & 1) << 4)) << 4) + ((kd1 & 7) | ((kd1 >> 4) << 3))] = (bf16_t)s1[r];
      }
    }
    asm volatile("s_wait_dscnt 0" ::: "memory");   // lane-crossing via LDS, same wave

    for (int nt = 0; nt < 4; ++nt)
      for (int r = 0; r < 8; ++r)
        accO[nt][r] *= scale[r];
    v16bf pf = load_frag(pw, lane);
    for (int nt = 0; nt < 4; ++nt)
      accO[nt] = wmma_bf16(pf, load_frag(&ldsV[nt << 9], lane), accO[nt]);
  }

  // normalize and store a (bf16, [ng][64][2304], transposed from C-layout)
  for (int nt = 0; nt < 4; ++nt)
    for (int r = 0; r < 8; ++r) {
      int m = r + ((lane >> 4) << 3);
      float val = accO[nt][r] / rsum[r];
      aout[((size_t)ng * CG + nt * 16 + (lane & 15)) * HWP + qt * 16 + m] = (bf16_t)val;
    }
}

// ---------------------------------------------------------------------------
// Fused gates + LSTM update: per (n,g,16-pixel tile), 4 waves each own a
// 16-channel tile and compute all four gate 1x1 GEMMs (WMMA, K=64), add the
// staged conv3x3 partials + bias, apply activations, update cell, emit h_new.
// The 64x16 a-tile (row stride 2304) is DMA'd in by the Tensor Data Mover.
// ---------------------------------------------------------------------------
__global__ void __launch_bounds__(128) gates_kernel(
    const bf16_t* __restrict__ a, const bf16_t* __restrict__ packWa,
    const float* __restrict__ gxi, const float* __restrict__ gxf,
    const float* __restrict__ gxg, const float* __restrict__ gxo,
    const float* __restrict__ bi, const float* __restrict__ bfv,
    const float* __restrict__ bgv, const float* __restrict__ bo,
    const float* __restrict__ cin, float* __restrict__ hout) {
  __shared__ __align__(32) bf16_t rawA[1024];    // 64 c-rows x 16 q (TDM dest)
  __shared__ __align__(32) bf16_t ldsB[1024];
  int lane = threadIdx.x & 31, wave = threadIdx.x >> 5;   // 4 waves
  int tq = blockIdx.x;                  // 16-pixel tile (0..143)
  int ng = blockIdx.y;                  // n*G + g
  int nb = ng >> 3, g = ng & 7;
  if (wave == 0) {
    tdm_load_2d(a + (size_t)ng * CG * HWP + tq * 16, LDS_OFF(rawA),
                HWP - tq * 16, HWP, CG, 16);
    __builtin_amdgcn_s_wait_tensorcnt(0);
  }
  __syncthreads();
  for (int e = threadIdx.x; e < 1024; e += 128) {
    int ksx = e >> 9, ee = e & 511;
    int k = ee >> 4, nn = ee & 15;
    int c = ksx * 32 + k;
    ldsB[(ksx << 9) + frag_pos(k, nn)] = rawA[(c << 4) + nn];
  }
  __syncthreads();
  int mt = wave;
  v16bf b0 = load_frag(&ldsB[0], lane);
  v16bf b1 = load_frag(&ldsB[512], lane);
  v8f acc[4] = {};
  for (int gate = 0; gate < 4; ++gate) {
    const bf16_t* wp = packWa + ((gate * 64 + (g * 4 + mt) * 2) << 9);
    acc[gate] = wmma_bf16(load_frag(wp, lane), b0, acc[gate]);
    acc[gate] = wmma_bf16(load_frag(wp + 512, lane), b1, acc[gate]);
  }
  int q = tq * 16 + (lane & 15);
  int rbase = mt * 16 + ((lane >> 4) << 3);
  for (int r = 0; r < 8; ++r) {
    int m   = rbase + r;
    int chn = g * CG + m;
    size_t off = (size_t)(ng * CG + m) * HWP + q;
    float iv = acc[0][r] + gxi[off] + bi[chn];
    float fv = acc[1][r] + gxf[off] + bfv[chn];
    float gv = acc[2][r] + gxg[off] + bgv[chn];
    float ov = acc[3][r] + gxo[off] + bo[chn];
    float ig = 1.f / (1.f + __expf(-iv));
    float fg = 1.f / (1.f + __expf(-fv));
    float gg = tanhf(gv);
    float og = 1.f / (1.f + __expf(-ov));
    size_t go = (size_t)(nb * CCH + chn) * HWP + q;
    float cn = fg * cin[go] + ig * gg;
    hout[go] = og * tanhf(cn);
  }
}

// ---------------------------------------------------------------------------
extern "C" void kernel_launch(void* const* d_in, const int* in_sizes, int n_in,
                              void* d_out, int out_size, void* d_ws, size_t ws_size,
                              hipStream_t stream) {
  (void)in_sizes; (void)n_in; (void)out_size; (void)ws_size;
  const float* x_in = (const float*)d_in[0];
  const float* h    = (const float*)d_in[1];
  const float* cst  = (const float*)d_in[2];
  const float* tau  = (const float*)d_in[3];
  const float* W_x  = (const float*)d_in[4];
  const float* W_ig = (const float*)d_in[5];
  const float* Wqkv[3] = {(const float*)d_in[6], (const float*)d_in[7], (const float*)d_in[8]};
  const float* Wa4[4]  = {(const float*)d_in[9],  (const float*)d_in[12],
                          (const float*)d_in[15], (const float*)d_in[18]};
  const float* Wx4[4]  = {(const float*)d_in[10], (const float*)d_in[13],
                          (const float*)d_in[16], (const float*)d_in[19]};
  const float* bia[4]  = {(const float*)d_in[11], (const float*)d_in[14],
                          (const float*)d_in[17], (const float*)d_in[20]};

  char* ws = (char*)d_ws;
  size_t off = 0;
  auto take = [&](size_t bytes) -> char* {
    char* p = ws + off;
    off = (off + bytes + 255) & ~(size_t)255;
    return p;
  };
  bf16_t* pWcat = (bf16_t*)take((size_t)32 * KSP * 512 * 2);
  bf16_t* pWqkv[3];
  for (int i = 0; i < 3; ++i) pWqkv[i] = (bf16_t*)take((size_t)32 * KS3 * 512 * 2);
  bf16_t* pWx[4];
  for (int i = 0; i < 4; ++i) pWx[i] = (bf16_t*)take((size_t)32 * KS3 * 512 * 2);
  bf16_t* pWa = (bf16_t*)take((size_t)4 * 64 * 512 * 2);
  bf16_t* xh  = (bf16_t*)take((size_t)NB * 1024 * HWP * 2);
  float*  qf  = (float*)take((size_t)16 * CG * HWP * 4);
  float*  kf  = (float*)take((size_t)16 * CG * DKV * 4);
  float*  vf  = (float*)take((size_t)16 * CG * DKV * 4);
  float*  gx[4];
  for (int i = 0; i < 4; ++i) gx[i] = (float*)take((size_t)16 * CG * HWP * 4);
  bf16_t* qT  = (bf16_t*)take((size_t)16 * HWP * CG * 2);
  bf16_t* kbm = (bf16_t*)take((size_t)16 * CG * DKV * 2);
  bf16_t* vbm = (bf16_t*)take((size_t)16 * CG * DKV * 2);
  bf16_t* abf = (bf16_t*)take((size_t)16 * CG * HWP * 2);

  auto pack = [&](const float* W, bf16_t* out, int M, int Ksrc, int stride,
                  int kOff, int Ktot, int rpg) {
    int total = M * Ksrc;
    pack_a_kernel<<<(total + 255) / 256, 256, 0, stream>>>(W, out, M, Ksrc,
                                                           stride, kOff, Ktot, rpg);
  };
  // weight pre-packing into WMMA fragment order
  pack(W_x, pWcat, 512, 256, 256, 0, 768, 512);
  pack(W_ig, pWcat, 512, 512, 512, 256, 768, 512);
  for (int i = 0; i < 3; ++i) pack(Wqkv[i], pWqkv[i], 512, K9, K9, 0, K9, 64);
  for (int i = 0; i < 4; ++i) pack(Wx4[i], pWx[i], 512, K9, K9, 0, K9, 64);
  for (int i = 0; i < 4; ++i) pack(Wa4[i], pWa + (size_t)i * 32768, 512, 64, 64, 0, 64, 64);

  // build interleaved xh: h half + projected x half (WMMA GEMM)
  h_to_xh_kernel<<<(NB * CCH * HWP + 255) / 256, 256, 0, stream>>>(h, xh);
  proj_gemm_kernel<<<dim3(HWP / 16, 4, NB), 256, 0, stream>>>(x_in, h, pWcat, xh);

  // grouped 3x3 convs (implicit GEMM, WMMA)
  conv3x3_gemm_kernel<<<dim3(HWP / 32, 16), 256, 0, stream>>>(xh, pWqkv[0], qf, 1, WW, HWP);
  conv3x3_gemm_kernel<<<dim3((DKV + 31) / 32, 16), 256, 0, stream>>>(xh, pWqkv[1], kf, 0, ODIM, DKV);
  conv3x3_gemm_kernel<<<dim3((DKV + 31) / 32, 16), 256, 0, stream>>>(xh, pWqkv[2], vf, 0, ODIM, DKV);
  for (int i = 0; i < 4; ++i)
    conv3x3_gemm_kernel<<<dim3(HWP / 32, 16), 256, 0, stream>>>(xh, pWx[i], gx[i], 1, WW, HWP);

  // attention operand repacks
  repack_qT_kernel<<<(16 * CG * HWP + 255) / 256, 256, 0, stream>>>(qf, qT);
  cvt_f32_bf16_kernel<<<(16 * CG * DKV + 255) / 256, 256, 0, stream>>>(kf, kbm, 16 * CG * DKV);
  cvt_f32_bf16_kernel<<<(16 * CG * DKV + 255) / 256, 256, 0, stream>>>(vf, vbm, 16 * CG * DKV);

  // flash attention with TDM-staged K/V (never materializes the 312MB logits)
  attention_kernel<<<dim3(HWP / 16 / 8, 16), 256, 0, stream>>>(qT, kbm, vbm, tau, abf);

  // fused gate GEMMs (TDM-staged a-tiles) + LSTM pointwise update -> h_new
  gates_kernel<<<dim3(HWP / 16, 16), 128, 0, stream>>>(
      abf, pWa, gx[0], gx[1], gx[2], gx[3], bia[0], bia[1], bia[2], bia[3],
      cst, (float*)d_out);
}
